// SPROUT_BrainLike_30571577213823
// MI455X (gfx1250) — matrix operand order repaired and checked by reading, not compiled, not run on hardware
//
#include <hip/hip_runtime.h>
#include <hip/hip_bf16.h>
#include <math.h>

// ---------------------------------------------------------------------------
// Model constants
// ---------------------------------------------------------------------------
#define BB 32      // batch
#define SS 512     // sequence length
#define DD 256     // model dim
#define NHD 4      // heads
#define HDD 64     // head dim
#define NN 4096    // neurons
#define VV 30000   // vocab
#define K0A 128    // active neurons
#define NSTEPS 5

typedef __attribute__((ext_vector_type(16))) _Float16 v16h;
typedef __attribute__((ext_vector_type(8)))  float    v8f;

union FragH { v16h v; float4 q[2]; };

// ---------------------------------------------------------------------------
// WMMA fragment helpers (CDNA5 v_wmma_f32_16x16x32_f16 layouts, ISA 7.12.2)
//  A (16x32, MxK) row-major, K contiguous:
//    lanes 0-15 : row=lane,   elems 0-7 = K k0..k0+7,  elems 8-15 = K k0+16..k0+23
//    lanes 16-31: row=lane-16, same with k0+8 base
//  B (32x16, KxN): column n = row n of W[N,K] (K contiguous):
//    lanes 0-15 : col=lane,   elems 0-7 = K 0..7,  elems 8-15 = K 16..23
//    lanes 16-31: col=lane-16, same with K 8 base
//  C/D (16x16 f32): lane n=lane&15, vgpr v -> row v + (lane&16 ? 8 : 0)
// ---------------------------------------------------------------------------
__device__ __forceinline__ v16h load_fragA(const _Float16* base, int ld) {
  int lane = threadIdx.x & 31;
  const _Float16* p = base + (size_t)(lane & 15) * ld + ((lane & 16) ? 8 : 0);
  FragH f;
  f.q[0] = *reinterpret_cast<const float4*>(p);
  f.q[1] = *reinterpret_cast<const float4*>(p + 16);
  return f.v;
}
__device__ __forceinline__ v16h load_fragB(const _Float16* base, int ld) {
  int lane = threadIdx.x & 31;
  const _Float16* p = base + (size_t)(lane & 15) * ld + ((lane & 16) ? 8 : 0);
  FragH f;
  f.q[0] = *reinterpret_cast<const float4*>(p);
  f.q[1] = *reinterpret_cast<const float4*>(p + 16);
  return f.v;
}
__device__ __forceinline__ v8f wmma32(v16h a, v16h b, v8f c) {
  return __builtin_amdgcn_wmma_f32_16x16x32_f16(false, a, false, b, (short)0, c,
                                                false, false);
}

template<int ACT>
__device__ __forceinline__ float act_apply(float x) {
  if (ACT == 1) return x > 0.f ? x : 0.f;                              // relu
  if (ACT == 2) return 0.5f * x * (1.f + erff(x * 0.70710678118f));    // gelu
  if (ACT == 3) return 1.f / (1.f + expf(-x));                        // sigmoid
  return x;
}

// ---------------------------------------------------------------------------
// Generic GEMM: C[M,N] = act(A[M,K] @ W[N,K]^T + bias)
// 4 waves/block; each wave computes a 32x64 strip: 2 A-fragments x 4
// B-fragments -> 8 WMMAs per k-step. Out-of-range N tiles are CLAMPED onto
// the last valid tile so the k-loop is fully uniform (no accumulator phis);
// the clamped duplicates are dropped in the epilogue. Requires M % 32 == 0,
// N % 16 == 0.
// ---------------------------------------------------------------------------
template<int ACT, bool O32, bool O16>
__global__ __launch_bounds__(128)
void k_gemm(const _Float16* __restrict__ A, int lda,
            const _Float16* __restrict__ W, int ldw,
            const float* __restrict__ bias,
            float* __restrict__ C32, _Float16* __restrict__ C16,
            int ldc, int N, int K) {
  const int wave = threadIdx.x >> 5;
  const int m0 = blockIdx.x * 32;
  const int n0 = (blockIdx.y * 16 + wave * 4) * 16;

  const _Float16* Wp[4];
#pragma unroll
  for (int t = 0; t < 4; ++t) {
    int n = n0 + t * 16;
    Wp[t] = W + (size_t)((n < N) ? n : (N - 16)) * ldw;  // clamp: keeps flow uniform
  }
  const _Float16* A0 = A + (size_t)m0 * lda;
  const _Float16* A1 = A0 + (size_t)16 * lda;

  v8f acc0[4] = {};
  v8f acc1[4] = {};
  for (int k = 0; k < K; k += 32) {
    v16h a0 = load_fragA(A0 + k, lda);
    v16h a1 = load_fragA(A1 + k, lda);
#pragma unroll
    for (int t = 0; t < 4; ++t) {
      const _Float16* Wb = Wp[t] + k;
      __builtin_prefetch(Wb + 32, 0, 0);  // speculative prefetch of next k-step
      v16h bf = load_fragB(Wb, ldw);
      acc0[t] = wmma32(a0, bf, acc0[t]);
      acc1[t] = wmma32(a1, bf, acc1[t]);
    }
  }

  const int lane = threadIdx.x & 31;
  const int nc = lane & 15;
  const int mb = (lane & 16) ? 8 : 0;
#pragma unroll
  for (int t = 0; t < 4; ++t) {
    if (n0 + t * 16 < N) {
      int col = n0 + t * 16 + nc;
      float bv = bias ? bias[col] : 0.f;
#pragma unroll
      for (int v = 0; v < 8; ++v) {
        float x0 = act_apply<ACT>(acc0[t][v] + bv);
        float x1 = act_apply<ACT>(acc1[t][v] + bv);
        size_t o0 = (size_t)(m0 + mb + v) * ldc + col;
        size_t o1 = (size_t)(m0 + 16 + mb + v) * ldc + col;
        if (O32) { C32[o0] = x0; C32[o1] = x1; }
        if (O16) { C16[o0] = (_Float16)x0; C16[o1] = (_Float16)x1; }
      }
    }
  }
}

template<int ACT, bool O32, bool O16>
static inline void launch_gemm(hipStream_t st, const _Float16* A, int lda,
                               const _Float16* W, int ldw, const float* bias,
                               float* C32, _Float16* C16, int ldc,
                               int M, int N, int K) {
  dim3 grid(M / 32, (((N + 15) / 16) + 15) / 16);
  k_gemm<ACT, O32, O16><<<grid, 128, 0, st>>>(A, lda, W, ldw, bias, C32, C16,
                                              ldc, N, K);
}

// ---------------------------------------------------------------------------
// Self attention: one wave per (batch, head, 16-row q tile). Scores via WMMA
// into LDS, serial softmax (per-row lanes), context via WMMA with A from LDS.
// LK = sequence length (512 encoder / 128 interaction). MASK: key mask av>0.01
// ---------------------------------------------------------------------------
template<int LK, bool MASK>
__global__ __launch_bounds__(32)
void k_attn(const _Float16* __restrict__ qkv, int ldq, int qoff, int koff,
            const _Float16* __restrict__ vT,       // [B, NH, HDD, LK]
            _Float16* __restrict__ outp, int ldo,  // write at col h*64
            const float* __restrict__ av) {
  const int qt = blockIdx.x, h = blockIdx.y, b = blockIdx.z;
  __shared__ __align__(16) float    sc[16 * LK];
  __shared__ __align__(16) _Float16 pr[16 * LK];
  const int lane = threadIdx.x;
  const _Float16* Aq = qkv + (size_t)(b * LK + qt * 16) * ldq + qoff + h * HDD;

  for (int kt = 0; kt < LK / 16; ++kt) {
    const _Float16* Bk = qkv + (size_t)(b * LK + kt * 16) * ldq + koff + h * HDD;
    v8f acc = {};
#pragma unroll
    for (int k = 0; k < HDD; k += 32) {
      v16h a  = load_fragA(Aq + k, ldq);
      v16h bf = load_fragB(Bk + k, ldq);
      acc = wmma32(a, bf, acc);
    }
    int n = kt * 16 + (lane & 15);
    int mb = (lane & 16) ? 8 : 0;
#pragma unroll
    for (int v = 0; v < 8; ++v) sc[(mb + v) * LK + n] = acc[v] * 0.125f;
  }
  __syncthreads();
  if (lane < 16) {
    float mx = -1e30f;
    for (int s = 0; s < LK; ++s) {
      float x = sc[lane * LK + s];
      if (MASK && !(av[b * LK + s] > 0.01f)) { x = -1e9f; sc[lane * LK + s] = x; }
      mx = fmaxf(mx, x);
    }
    float sum = 0.f;
    for (int s = 0; s < LK; ++s) {
      float e = expf(sc[lane * LK + s] - mx);
      sc[lane * LK + s] = e; sum += e;
    }
    float inv = 1.f / sum;
    for (int s = 0; s < LK; ++s) pr[lane * LK + s] = (_Float16)(sc[lane * LK + s] * inv);
  }
  __syncthreads();
  const _Float16* Vt = vT + ((size_t)(b * NHD + h) * HDD) * LK;
#pragma unroll
  for (int nt = 0; nt < 4; ++nt) {
    v8f acc = {};
    for (int k = 0; k < LK; k += 32) {
      v16h a  = load_fragA(pr + k, LK);
      v16h bf = load_fragB(Vt + (size_t)(nt * 16) * LK + k, LK);
      acc = wmma32(a, bf, acc);
    }
    int n = lane & 15, mb = (lane & 16) ? 8 : 0;
#pragma unroll
    for (int v = 0; v < 8; ++v)
      outp[(size_t)(b * LK + qt * 16 + mb + v) * ldo + h * HDD + nt * 16 + n] =
          (_Float16)acc[v];
  }
}

// ---------------------------------------------------------------------------
// Small elementwise / reduction kernels
// ---------------------------------------------------------------------------
__global__ void k_cvt(const float* __restrict__ s, _Float16* __restrict__ d, int n) {
  int e = blockIdx.x * blockDim.x + threadIdx.x;
  if (e < n) d[e] = (_Float16)s[e];
}

__global__ void k_embed(const int* __restrict__ tok, const float* __restrict__ te,
                        const float* __restrict__ pe, float* __restrict__ x32,
                        _Float16* __restrict__ x16) {
  int e = blockIdx.x * blockDim.x + threadIdx.x;
  if (e >= BB * SS * DD) return;
  int d = e & 255, s = (e >> 8) & 511, b = e >> 17;
  int t = tok[b * SS + s];
  float v = te[(size_t)t * DD + d] + pe[s * DD + d];
  x32[e] = v; x16[e] = (_Float16)v;
}

__global__ void k_vtrans(const _Float16* __restrict__ qkv, int ldq, int voff,
                         _Float16* __restrict__ vT, int L, int total) {
  int e = blockIdx.x * blockDim.x + threadIdx.x;
  if (e >= total) return;
  int s = e % L, d = (e / L) % HDD, h = (e / (L * HDD)) % NHD, b = e / (L * HDD * NHD);
  vT[e] = qkv[(size_t)(b * L + s) * ldq + voff + h * HDD + d];
}

__global__ __launch_bounds__(256)
void k_add_ln(const float* __restrict__ x, const float* __restrict__ add,
              const float* __restrict__ g, const float* __restrict__ bta,
              float* __restrict__ y32, _Float16* __restrict__ y16) {
  __shared__ float red[DD];
  int row = blockIdx.x, t = threadIdx.x;
  float v = x[(size_t)row * DD + t] + (add ? add[(size_t)row * DD + t] : 0.f);
  red[t] = v; __syncthreads();
  for (int o = 128; o > 0; o >>= 1) { if (t < o) red[t] += red[t + o]; __syncthreads(); }
  float mu = red[0] / DD; __syncthreads();
  float c = v - mu;
  red[t] = c * c; __syncthreads();
  for (int o = 128; o > 0; o >>= 1) { if (t < o) red[t] += red[t + o]; __syncthreads(); }
  float y = c * rsqrtf(red[0] / DD + 1e-5f) * g[t] + bta[t];
  if (y32) y32[(size_t)row * DD + t] = y;
  if (y16) y16[(size_t)row * DD + t] = (_Float16)y;
}

__global__ void k_gru(const float* __restrict__ gi, const float* __restrict__ gh,
                      const float* __restrict__ hprev, float* __restrict__ nh,
                      int total) {
  int e = blockIdx.x * blockDim.x + threadIdx.x;
  if (e >= total) return;
  int r = e >> 8, d = e & 255;
  float ir = gi[(size_t)r * 768 + d],       hr = gh[(size_t)r * 768 + d];
  float iz = gi[(size_t)r * 768 + 256 + d], hz = gh[(size_t)r * 768 + 256 + d];
  float in = gi[(size_t)r * 768 + 512 + d], hn = gh[(size_t)r * 768 + 512 + d];
  float rr = 1.f / (1.f + expf(-(ir + hr)));
  float zz = 1.f / (1.f + expf(-(iz + hz)));
  float n  = tanhf(in + rr * hn);
  nh[e] = (1.f - zz) * n + zz * hprev[e];
}

__global__ void k_concat(const _Float16* __restrict__ a, const _Float16* __restrict__ b,
                         _Float16* __restrict__ c, int total) {
  int e = blockIdx.x * blockDim.x + threadIdx.x;
  if (e >= total) return;
  int r = e >> 9, d = e & 511;
  c[e] = (d < 256) ? a[(size_t)r * 256 + d] : b[(size_t)r * 256 + d - 256];
}

__global__ void k_scalerows(const float* __restrict__ m, const float* __restrict__ av,
                            _Float16* __restrict__ o, int total) {
  int e = blockIdx.x * blockDim.x + threadIdx.x;
  if (e < total) o[e] = (_Float16)(m[e] * av[e >> 8]);
}

__global__ __launch_bounds__(256)
void k_delta(const _Float16* __restrict__ u1, const float* __restrict__ w2,
             const float* __restrict__ b2, float* __restrict__ av,
             float* __restrict__ msk) {
  __shared__ float red[256];
  int row = blockIdx.x, t = threadIdx.x;
  red[t] = (float)u1[(size_t)row * 256 + t] * w2[t];
  __syncthreads();
  for (int o = 128; o > 0; o >>= 1) { if (t < o) red[t] += red[t + o]; __syncthreads(); }
  if (t == 0) {
    float del = 1.f / (1.f + expf(-(red[0] + b2[0])));
    float a0 = av[row];
    float na = fminf(fmaxf(0.7f * a0 + 0.3f * del, 0.f), 1.f);
    float m = (a0 > 0.01f) ? 1.f : 0.f;
    av[row] = na * m;
    msk[row] = m;
  }
}

__global__ void k_hidupd(const float* __restrict__ nh, const float* __restrict__ msk,
                         float* __restrict__ h32, _Float16* __restrict__ h16, int total) {
  int e = blockIdx.x * blockDim.x + threadIdx.x;
  if (e >= total) return;
  float v = nh[e] * msk[e >> 8];
  h32[e] = v; h16[e] = (_Float16)v;
}

__global__ void k_hidinit(const float* __restrict__ sig, const int* __restrict__ idx,
                          const float* __restrict__ av, float* __restrict__ h32,
                          _Float16* __restrict__ h16) {
  int e = blockIdx.x * blockDim.x + threadIdx.x;
  if (e >= BB * K0A * DD) return;
  int d = e & 255, j = (e >> 8) & 127, b = e >> 15;
  float a = av[b * K0A + j];
  float v = sig[(size_t)idx[b * K0A + j] * DD + d] * a;
  h32[e] = v; h16[e] = (_Float16)v;
}

__global__ __launch_bounds__(256)
void k_topk(const float* __restrict__ logits, int* __restrict__ idx,
            float* __restrict__ av) {
  __shared__ float vals[NN];
  __shared__ float rv[256];
  __shared__ int   ri[256];
  int b = blockIdx.x, t = threadIdx.x;
  for (int i = t; i < NN; i += 256) vals[i] = logits[(size_t)b * NN + i];
  __syncthreads();
  for (int j = 0; j < K0A; ++j) {
    float bv = -3.0e38f; int bi = 0;
    for (int i = t; i < NN; i += 256) { float x = vals[i]; if (x > bv) { bv = x; bi = i; } }
    rv[t] = bv; ri[t] = bi; __syncthreads();
    for (int o = 128; o > 0; o >>= 1) {
      if (t < o && rv[t + o] > rv[t]) { rv[t] = rv[t + o]; ri[t] = ri[t + o]; }
      __syncthreads();
    }
    if (t == 0) {
      idx[b * K0A + j] = ri[0];
      av[b * K0A + j] = 1.f / (1.f + expf(-rv[0]));
      vals[ri[0]] = -3.0e38f;
    }
    __syncthreads();
  }
}

__global__ __launch_bounds__(256)
void k_poolq(const float* __restrict__ pq, const float* __restrict__ pWi,
             const float* __restrict__ pbi, float* __restrict__ qvec) {
  int j = threadIdx.x;
  float s = pbi[j];
  for (int d = 0; d < DD; ++d) s += pq[d] * pWi[(size_t)j * DD + d];
  qvec[j] = s;
}

__global__ __launch_bounds__(128)
void k_poolattn(const float* __restrict__ qvec, const _Float16* __restrict__ kv,
                float* __restrict__ pooled, _Float16* __restrict__ pooled16) {
  int h = blockIdx.x, b = blockIdx.y, t = threadIdx.x;
  __shared__ float sc[SS];
  __shared__ float red[128];
  for (int s = t; s < SS; s += 128) {
    const _Float16* kp = kv + (size_t)(b * SS + s) * 512 + h * HDD;
    float dot = 0.f;
    for (int d = 0; d < HDD; ++d) dot += qvec[h * HDD + d] * (float)kp[d];
    sc[s] = dot * 0.125f;
  }
  __syncthreads();
  float lm = -1e30f;
  for (int s = t; s < SS; s += 128) lm = fmaxf(lm, sc[s]);
  red[t] = lm; __syncthreads();
  for (int o = 64; o > 0; o >>= 1) { if (t < o) red[t] = fmaxf(red[t], red[t + o]); __syncthreads(); }
  float mx = red[0]; __syncthreads();
  float ls = 0.f;
  for (int s = t; s < SS; s += 128) { float e = expf(sc[s] - mx); sc[s] = e; ls += e; }
  red[t] = ls; __syncthreads();
  for (int o = 64; o > 0; o >>= 1) { if (t < o) red[t] += red[t + o]; __syncthreads(); }
  float inv = 1.f / red[0];
  __syncthreads();
  if (t < HDD) {
    float o = 0.f;
    for (int s = 0; s < SS; ++s)
      o += sc[s] * inv * (float)kv[(size_t)(b * SS + s) * 512 + 256 + h * HDD + t];
    pooled[b * DD + h * HDD + t] = o;
    pooled16[b * DD + h * HDD + t] = (_Float16)o;
  }
}

__global__ void k_zero2(float* __restrict__ a, _Float16* __restrict__ b, int n) {
  int e = blockIdx.x * blockDim.x + threadIdx.x;
  if (e < n) { a[e] = 0.f; b[e] = (_Float16)0.f; }
}

__global__ void k_scatter(const int* __restrict__ idx, const float* __restrict__ av,
                          float* __restrict__ a32, _Float16* __restrict__ a16) {
  int e = blockIdx.x * blockDim.x + threadIdx.x;
  if (e >= BB * K0A) return;
  int b = e >> 7;
  int i = idx[e];
  float v = av[e];
  a32[(size_t)b * NN + i] = v;
  a16[(size_t)b * NN + i] = (_Float16)v;
}

// ---------------------------------------------------------------------------
// Input indices: jax pytree (sorted-key) flattening of {'params':..., 'tokens':...}
// ---------------------------------------------------------------------------
enum {
  P_aW1 = 0, P_aW2, P_ab1, P_ab2, P_dW1, P_dW2, P_db1, P_db2, P_dlnb, P_dlng,
  P_ENC0 = 10,  // aWi,aWo,abi,abo,fW1,fW2,fb1,fb2,ln1b,ln1g,ln2b,ln2g
  P_ENC1 = 22,
  P_gWhh = 34, P_gWih, P_gbhh, P_gbih, P_iWi, P_iWo, P_ibi, P_ibo, P_ilnb, P_ilng,
  P_oW, P_ob, P_pWi, P_pWo, P_pbi, P_pbo, P_pos, P_pq, P_sig, P_tok, P_uW1, P_uW2,
  P_ub1, P_ub2, P_tokens
};
// enc sub-order offsets:
enum { E_aWi = 0, E_aWo, E_abi, E_abo, E_fW1, E_fW2, E_fb1, E_fb2, E_ln1b, E_ln1g,
       E_ln2b, E_ln2g };

static inline dim3 ew(size_t n) { return dim3((unsigned)((n + 255) / 256)); }

extern "C" void kernel_launch(void* const* d_in, const int* in_sizes, int n_in,
                              void* d_out, int out_size, void* d_ws, size_t ws_size,
                              hipStream_t stream) {
  (void)in_sizes; (void)n_in; (void)out_size; (void)ws_size;
  auto F = [&](int i) { return (const float*)d_in[i]; };
  const int* tokens = (const int*)d_in[P_tokens];

  char* ws = (char*)d_ws;
  size_t off = 0;
  auto alloc  = [&](size_t bytes) -> void* {
    void* p = ws + off; off += (bytes + 255) & ~(size_t)255; return p;
  };
  auto aH = [&](size_t n) { return (_Float16*)alloc(n * 2); };
  auto aF = [&](size_t n) { return (float*)alloc(n * 4); };

  // ---- f16 weight conversions -------------------------------------------
  _Float16* w_aW1 = aH(512 * 256);
  _Float16* w_aW2 = aH(4096 * 512);
  _Float16* w_dW1 = aH(512 * 4096);
  _Float16* w_dW2 = aH(256 * 512);
  _Float16* w_oW  = aH((size_t)VV * 256);
  _Float16* w_iWi = aH(768 * 256);
  _Float16* w_iWo = aH(256 * 256);
  _Float16* w_gWih = aH(768 * 256);
  _Float16* w_gWhh = aH(768 * 256);
  _Float16* w_uW1 = aH(256 * 512);
  _Float16* w_pWi = aH(768 * 256);
  _Float16* w_pWo = aH(256 * 256);
  _Float16* w_eaWi[2] = { aH(768 * 256), aH(768 * 256) };
  _Float16* w_eaWo[2] = { aH(256 * 256), aH(256 * 256) };
  _Float16* w_efW1[2] = { aH(1024 * 256), aH(1024 * 256) };
  _Float16* w_efW2[2] = { aH(256 * 1024), aH(256 * 1024) };

  auto cvt = [&](int pi, _Float16* dst, size_t n) {
    k_cvt<<<ew(n), 256, 0, stream>>>(F(pi), dst, (int)n);
  };
  cvt(P_aW1, w_aW1, 512 * 256);
  cvt(P_aW2, w_aW2, 4096 * 512);
  cvt(P_dW1, w_dW1, 512 * 4096);
  cvt(P_dW2, w_dW2, 256 * 512);
  cvt(P_oW,  w_oW,  (size_t)VV * 256);
  cvt(P_iWi, w_iWi, 768 * 256);
  cvt(P_iWo, w_iWo, 256 * 256);
  cvt(P_gWih, w_gWih, 768 * 256);
  cvt(P_gWhh, w_gWhh, 768 * 256);
  cvt(P_uW1, w_uW1, 256 * 512);
  cvt(P_pWi, w_pWi, 768 * 256);
  cvt(P_pWo, w_pWo, 256 * 256);
  for (int L = 0; L < 2; ++L) {
    int base = (L == 0) ? P_ENC0 : P_ENC1;
    cvt(base + E_aWi, w_eaWi[L], 768 * 256);
    cvt(base + E_aWo, w_eaWo[L], 256 * 256);
    cvt(base + E_fW1, w_efW1[L], 1024 * 256);
    cvt(base + E_fW2, w_efW2[L], 256 * 1024);
  }

  // ---- activations -------------------------------------------------------
  const int TOK = BB * SS;            // 16384
  const int AR  = BB * K0A;           // 4096 active rows
  float*    x32    = aF((size_t)TOK * DD);
  _Float16* x16    = aH((size_t)TOK * DD);
  _Float16* qkv16  = aH((size_t)TOK * 768);     // also reused: kv16, iqkv16
  _Float16* vT     = aH((size_t)BB * NHD * HDD * SS); // also reused for steps
  _Float16* abuf16 = aH((size_t)TOK * DD);      // attn out (also iattn16)
  float*    tmp32  = aF((size_t)TOK * DD);      // residual branch (also msgs32)
  _Float16* ffn1   = aH((size_t)TOK * 1024);
  float*    qvec   = aF(DD);
  float*    pooled = aF(BB * DD);
  _Float16* pooled16 = aH(BB * DD);
  _Float16* pooled2  = aH(BB * DD);
  _Float16* h1a    = aH(BB * 512);
  float*    logits = aF(BB * NN);
  int*      idx    = (int*)alloc(AR * 4);
  float*    av     = aF(AR);
  float*    msk    = aF(AR);
  float*    hid32  = aF((size_t)AR * DD);
  _Float16* hid16  = aH((size_t)AR * DD);
  _Float16* msgs16 = aH((size_t)AR * DD);
  float*    gi32   = aF((size_t)AR * 768);
  float*    gh32   = aF((size_t)AR * 768);
  float*    nh32   = aF((size_t)AR * DD);
  _Float16* nh16   = aH((size_t)AR * DD);
  _Float16* comb16 = aH((size_t)AR * 512);
  _Float16* u116   = aH((size_t)AR * DD);
  float*    act32  = aF((size_t)BB * NN);
  _Float16* act16  = aH((size_t)BB * NN);
  _Float16* dh1    = aH(BB * 512);
  float*    d0     = aF(BB * DD);
  _Float16* d16    = aH(BB * DD);

  // ---- 1. Embedding ------------------------------------------------------
  k_embed<<<ew((size_t)TOK * DD), 256, 0, stream>>>(tokens, F(P_tok), F(P_pos),
                                                    x32, x16);

  // ---- 2. Encoder layers -------------------------------------------------
  for (int L = 0; L < 2; ++L) {
    int base = (L == 0) ? P_ENC0 : P_ENC1;
    // QKV projection: [TOK,256] @ [768,256]^T
    launch_gemm<0, false, true>(stream, x16, DD, w_eaWi[L], DD, F(base + E_abi),
                                nullptr, qkv16, 768, TOK, 768, DD);
    // V transpose -> [B,NH,HD,S]
    k_vtrans<<<ew((size_t)BB * NHD * HDD * SS), 256, 0, stream>>>(
        qkv16, 768, 512, vT, SS, BB * NHD * HDD * SS);
    // attention
    k_attn<SS, false><<<dim3(SS / 16, NHD, BB), 32, 0, stream>>>(
        qkv16, 768, 0, 256, vT, abuf16, DD, nullptr);
    // out-projection
    launch_gemm<0, true, false>(stream, abuf16, DD, w_eaWo[L], DD,
                                F(base + E_abo), tmp32, nullptr, DD, TOK, DD, DD);
    // residual + LN1
    k_add_ln<<<TOK, 256, 0, stream>>>(x32, tmp32, F(base + E_ln1g),
                                      F(base + E_ln1b), x32, x16);
    // FFN
    launch_gemm<1, false, true>(stream, x16, DD, w_efW1[L], DD, F(base + E_fb1),
                                nullptr, ffn1, 1024, TOK, 1024, DD);
    launch_gemm<0, true, false>(stream, ffn1, 1024, w_efW2[L], 1024,
                                F(base + E_fb2), tmp32, nullptr, DD, TOK, DD, 1024);
    k_add_ln<<<TOK, 256, 0, stream>>>(x32, tmp32, F(base + E_ln2g),
                                      F(base + E_ln2b), x32, x16);
  }

  // ---- 3. Attention pooling ---------------------------------------------
  _Float16* kv16 = qkv16;  // reuse: [TOK, 512] (k | v)
  launch_gemm<0, false, true>(stream, x16, DD, w_pWi + (size_t)256 * DD, DD,
                              F(P_pbi) + 256, nullptr, kv16, 512, TOK, 512, DD);
  k_poolq<<<1, 256, 0, stream>>>(F(P_pq), F(P_pWi), F(P_pbi), qvec);
  k_poolattn<<<dim3(NHD, BB), 128, 0, stream>>>(qvec, kv16, pooled, pooled16);
  launch_gemm<0, false, true>(stream, pooled16, DD, w_pWo, DD, F(P_pbo),
                              nullptr, pooled2, DD, BB, DD, DD);

  // ---- 4. Activation head + top-k ---------------------------------------
  launch_gemm<2, false, true>(stream, pooled2, DD, w_aW1, DD, F(P_ab1),
                              nullptr, h1a, 512, BB, 512, DD);
  launch_gemm<0, true, false>(stream, h1a, 512, w_aW2, 512, F(P_ab2),
                              logits, nullptr, NN, BB, NN, 512);
  k_topk<<<BB, 256, 0, stream>>>(logits, idx, av);
  k_hidinit<<<ew((size_t)AR * DD), 256, 0, stream>>>(F(P_sig), idx, av,
                                                     hid32, hid16);

  // ---- 5. Interaction steps ---------------------------------------------
  _Float16* iqkv  = qkv16;   // reuse: [AR, 768]
  _Float16* ivT   = vT;      // reuse: [B,NH,HD,128]
  _Float16* iattn = abuf16;  // reuse: [AR, 256]
  float*    msgs32 = tmp32;  // reuse: [AR, 256]
  for (int step = 0; step < NSTEPS; ++step) {
    launch_gemm<0, false, true>(stream, hid16, DD, w_iWi, DD, F(P_ibi),
                                nullptr, iqkv, 768, AR, 768, DD);
    k_vtrans<<<ew((size_t)BB * NHD * HDD * K0A), 256, 0, stream>>>(
        iqkv, 768, 512, ivT, K0A, BB * NHD * HDD * K0A);
    k_attn<K0A, true><<<dim3(K0A / 16, NHD, BB), 32, 0, stream>>>(
        iqkv, 768, 0, 256, ivT, iattn, DD, av);
    launch_gemm<0, true, false>(stream, iattn, DD, w_iWo, DD, F(P_ibo),
                                msgs32, nullptr, DD, AR, DD, DD);
    k_scalerows<<<ew((size_t)AR * DD), 256, 0, stream>>>(msgs32, av, msgs16,
                                                         AR * DD);
    launch_gemm<0, true, false>(stream, msgs16, DD, w_gWih, DD, F(P_gbih),
                                gi32, nullptr, 768, AR, 768, DD);
    launch_gemm<0, true, false>(stream, hid16, DD, w_gWhh, DD, F(P_gbhh),
                                gh32, nullptr, 768, AR, 768, DD);
    k_gru<<<ew((size_t)AR * DD), 256, 0, stream>>>(gi32, gh32, hid32, nh32,
                                                   AR * DD);
    k_add_ln<<<AR, 256, 0, stream>>>(nh32, nullptr, F(P_ilng), F(P_ilnb),
                                     nh32, nh16);
    k_concat<<<ew((size_t)AR * 512), 256, 0, stream>>>(hid16, nh16, comb16,
                                                       AR * 512);
    launch_gemm<2, false, true>(stream, comb16, 512, w_uW1, 512, F(P_ub1),
                                nullptr, u116, DD, AR, DD, 512);
    k_delta<<<AR, 256, 0, stream>>>(u116, F(P_uW2), F(P_ub2), av, msk);
    k_hidupd<<<ew((size_t)AR * DD), 256, 0, stream>>>(nh32, msk, hid32, hid16,
                                                      AR * DD);
  }

  // ---- 6. Scatter final activations -------------------------------------
  k_zero2<<<ew((size_t)BB * NN), 256, 0, stream>>>(act32, act16, BB * NN);
  k_scatter<<<ew(AR), 256, 0, stream>>>(idx, av, act32, act16);

  // ---- 7. Output decoder -------------------------------------------------
  launch_gemm<2, false, true>(stream, act16, NN, w_dW1, NN, F(P_db1),
                              nullptr, dh1, 512, BB, 512, NN);
  launch_gemm<0, true, false>(stream, dh1, 512, w_dW2, 512, F(P_db2),
                              d0, nullptr, DD, BB, DD, 512);
  k_add_ln<<<BB, 256, 0, stream>>>(d0, nullptr, F(P_dlng), F(P_dlnb),
                                   nullptr, d16);
  launch_gemm<0, true, false>(stream, d16, DD, w_oW, DD, F(P_ob),
                              (float*)d_out, nullptr, VV, BB, VV, DD);
}
